// DFNet_24017457119253
// MI455X (gfx1250) — matrix operand order repaired and checked by reading, not compiled, not run on hardware
//
#include <hip/hip_runtime.h>

// ---------------------------------------------------------------------------
// Tri-directional 3D Mamba block for MI455X (gfx1250, wave32).
//
// Roofline: ~4.2 GFLOP of GEMM, ~80 MB of intermediates (fully L2-resident on
// the 192 MB L2). GEMM time at v_wmma_f32_16x16x32_f16 rates is negligible;
// the binding constraint is the 4096-step sequential selective scan, which we
// parallelize across 768 channels x 16 states (one lane per state, shfl_xor
// tree for the per-step C-dot), i.e. 384 resident waves of dependent work.
// All matmuls with K>=128 use f16 WMMA with f32 accumulation; norms, conv,
// softplus/exp scan math, gelu and all epilogues are f32.
//
// GEMM waves compute a 16x64 tile (1 A fragment reused across 4 B fragments /
// 4 WMMAs per K-chunk) to raise arithmetic intensity; only the tiny N=48
// x_proj GEMM uses the single-tile path.
// ---------------------------------------------------------------------------

typedef __attribute__((ext_vector_type(16))) _Float16 v16h;
typedef __attribute__((ext_vector_type(8)))  _Float16 v8h;
typedef __attribute__((ext_vector_type(8)))  float    v8f;

enum { EP_F32 = 0, EP_CAT16 = 1, EP_BIAS_RES = 2, EP_BIAS_GELU = 3, EP_FINAL = 4 };

static __device__ __forceinline__ v16h joinh(v8h lo, v8h hi) {
  return __builtin_shufflevector(lo, hi, 0,1,2,3,4,5,6,7,8,9,10,11,12,13,14,15);
}

static __device__ __forceinline__ float wave_sum32(float v) {
  v += __shfl_xor(v, 16, 32);
  v += __shfl_xor(v, 8, 32);
  v += __shfl_xor(v, 4, 32);
  v += __shfl_xor(v, 2, 32);
  v += __shfl_xor(v, 1, 32);
  return v;
}

// Shared GEMM epilogue (uniform `mode` per launch).
static __device__ __forceinline__ void gemm_epilogue(
    int mode, v8f acc, int rbase, int col,
    float* __restrict__ outf, _Float16* __restrict__ outh, int ldo,
    const float* __restrict__ bias, const float* __restrict__ res,
    float* __restrict__ dout) {
  if (mode == EP_F32) {
    for (int r = 0; r < 8; ++r)
      outf[(size_t)(rbase + r) * ldo + col] = acc[r];
  } else if (mode == EP_CAT16) {                 // y -> cat[s*384 + n*128 + c], f16
    for (int r = 0; r < 8; ++r) {
      int t = rbase + r; int n = t >> 12; int s = t & 4095;
      outh[(size_t)s * 384 + n * 128 + col] = (_Float16)acc[r];
    }
  } else if (mode == EP_BIAS_RES) {              // + proj_b + x(ch-major) -> f32 (s,128)
    for (int r = 0; r < 8; ++r) {
      int row = rbase + r;
      outf[(size_t)row * ldo + col] = acc[r] + bias[col] + res[(size_t)col * 4096 + row];
    }
  } else if (mode == EP_BIAS_GELU) {             // + fc1_b, exact gelu -> f16
    for (int r = 0; r < 8; ++r) {
      float v = acc[r] + bias[col];
      outh[(size_t)(rbase + r) * ldo + col] =
          (_Float16)(0.5f * v * (1.f + erff(v * 0.70710678118654752f)));
    }
  } else {                                       // EP_FINAL: ch-major out + residual
    for (int r = 0; r < 8; ++r) {
      int row = rbase + r;
      dout[(size_t)col * 4096 + row] = acc[r] + bias[col] + res[(size_t)row * 128 + col];
    }
  }
}

// --------------------------- weight prep -----------------------------------
__global__ void k_f2h(const float* __restrict__ s, _Float16* __restrict__ d, int n) {
  int i = blockIdx.x * 256 + threadIdx.x;
  if (i < n) d[i] = (_Float16)s[i];
}

// x_proj_w is (40,256); pad to (48,256) so N is a multiple of 16 for WMMA.
__global__ void k_f2h_pad48(const float* __restrict__ s, _Float16* __restrict__ d) {
  int i = blockIdx.x * 256 + threadIdx.x;
  if (i < 48 * 256) {
    int r = i >> 8, c = i & 255;
    d[i] = (r < 40) ? (_Float16)s[r * 256 + c] : (_Float16)0.f;
  }
}

__global__ void k_negexp(const float* __restrict__ s, float* __restrict__ d, int n) {
  int i = blockIdx.x * 256 + threadIdx.x;
  if (i < n) d[i] = -__expf(s[i]);
}

// ------------------- fused LN(channel) + LN(channel) + 3-way permute --------
// x is (C=128, 4096) channel-major.  One wave per spatial location, 4 ch/lane.
__global__ void k_norm_seq(const float* __restrict__ x,
                           const float* __restrict__ lnw, const float* __restrict__ lnb,
                           const float* __restrict__ mw,  const float* __restrict__ mb,
                           _Float16* __restrict__ seq) {
  int wid = threadIdx.x >> 5, lane = threadIdx.x & 31;
  int s = blockIdx.x * 8 + wid;                 // 0..4095
  float v[4];
  float sum = 0.f;
  for (int j = 0; j < 4; ++j) {
    int c = lane * 4 + j;
    v[j] = x[(size_t)c * 4096 + s];
    sum += v[j];
  }
  float mean = wave_sum32(sum) * (1.f / 128.f);
  float var = 0.f;
  for (int j = 0; j < 4; ++j) { float dd = v[j] - mean; var += dd * dd; }
  var = wave_sum32(var) * (1.f / 128.f);
  float r = rsqrtf(var + 1e-6f);

  float x1[4]; float sum2 = 0.f;
  for (int j = 0; j < 4; ++j) {
    int c = lane * 4 + j;
    x1[j] = lnw[c] * ((v[j] - mean) * r) + lnb[c];
    sum2 += x1[j];
  }
  float mean2 = wave_sum32(sum2) * (1.f / 128.f);
  float var2 = 0.f;
  for (int j = 0; j < 4; ++j) { float dd = x1[j] - mean2; var2 += dd * dd; }
  var2 = wave_sum32(var2) * (1.f / 128.f);
  float r2 = rsqrtf(var2 + 1e-5f);

  int d = s >> 8, h = (s >> 4) & 15, w = s & 15;
  int l1 = h * 256 + w * 16 + d;                // H,W,D flatten
  int l2 = w * 256 + d * 16 + h;                // W,D,H flatten
  for (int j = 0; j < 4; ++j) {
    int c = lane * 4 + j;
    _Float16 oh = (_Float16)((x1[j] - mean2) * r2 * mw[c] + mb[c]);
    seq[(size_t)s * 128 + c]            = oh;
    seq[(size_t)(4096 + l1) * 128 + c]  = oh;
    seq[(size_t)(8192 + l2) * 128 + c]  = oh;
  }
}

// ---------------------- WMMA GEMM, 16x64 tile per wave ----------------------
// out(M,N) = A(M,K) * W(N,K)^T.  One A fragment feeds 4 WMMAs per K-chunk.
// Requires Nt % 4 == 0.
__global__ void k_gemm4(const _Float16* __restrict__ A, int lda,
                        const _Float16* __restrict__ W, int ldw,
                        int Mt, int Nt, int K,
                        float* __restrict__ outf, _Float16* __restrict__ outh, int ldo,
                        int mode,
                        const float* __restrict__ bias,
                        const float* __restrict__ res,
                        float* __restrict__ dout) {
  int wid = threadIdx.x >> 5, lane = threadIdx.x & 31;
  int Ng = Nt >> 2;
  int tile = blockIdx.x * 8 + wid;
  if (tile >= Mt * Ng) return;
  int ng = tile % Ng, mt = tile / Ng;
  int m0 = mt * 16, n0 = ng * 64;
  int hi = lane >> 4;                            // upper/lower half-wave
  int r16 = lane & 15;

  const _Float16* pa = A + (size_t)(m0 + r16) * lda + hi * 8;
  const _Float16* pb = W + (size_t)(n0 + r16) * ldw + hi * 8;
  const size_t bstep = (size_t)16 * ldw;         // next 16-column group of W

  v8f acc[4] = {{}, {}, {}, {}};
  for (int k0 = 0; k0 < K; k0 += 32) {
    __builtin_prefetch(pa + 64, 0, 3);           // global_prefetch_b8
    v8h al = *(const v8h*)(pa);
    v8h ah = *(const v8h*)(pa + 16);
    v16h a = joinh(al, ah);
    pa += 32;
#pragma unroll
    for (int j = 0; j < 4; ++j) {
      const _Float16* pbj = pb + (size_t)j * bstep;
      __builtin_prefetch(pbj + 64, 0, 3);
      v8h bl = *(const v8h*)(pbj);
      v8h bh = *(const v8h*)(pbj + 16);
      acc[j] = __builtin_amdgcn_wmma_f32_16x16x32_f16(
          false, a, false, joinh(bl, bh), (short)0, acc[j], false, false);
    }
    pb += 32;
  }

  int rbase = m0 + (hi ? 8 : 0);
#pragma unroll
  for (int j = 0; j < 4; ++j)
    gemm_epilogue(mode, acc[j], rbase, n0 + j * 16 + r16,
                  outf, outh, ldo, bias, res, dout);
}

// ------------------- WMMA GEMM, single 16x16 tile per wave ------------------
// Used only for the N=48 x_proj GEMM.
__global__ void k_gemm(const _Float16* __restrict__ A, int lda,
                       const _Float16* __restrict__ W, int ldw,
                       int Mt, int Nt, int K,
                       float* __restrict__ outf, _Float16* __restrict__ outh, int ldo,
                       int mode,
                       const float* __restrict__ bias,
                       const float* __restrict__ res,
                       float* __restrict__ dout) {
  int wid = threadIdx.x >> 5, lane = threadIdx.x & 31;
  int tile = blockIdx.x * 8 + wid;
  if (tile >= Mt * Nt) return;
  int nt = tile % Nt, mt = tile / Nt;
  int m0 = mt * 16, n0 = nt * 16;
  int hi = lane >> 4;
  int r16 = lane & 15;

  const _Float16* pa = A + (size_t)(m0 + r16) * lda + hi * 8;
  const _Float16* pb = W + (size_t)(n0 + r16) * ldw + hi * 8;

  v8f acc = {};
  for (int k0 = 0; k0 < K; k0 += 32) {
    __builtin_prefetch(pa + 64, 0, 3);
    __builtin_prefetch(pb + 64, 0, 3);
    v8h al = *(const v8h*)(pa);
    v8h ah = *(const v8h*)(pa + 16);
    v8h bl = *(const v8h*)(pb);
    v8h bh = *(const v8h*)(pb + 16);
    pa += 32; pb += 32;
    acc = __builtin_amdgcn_wmma_f32_16x16x32_f16(
        false, joinh(al, ah), false, joinh(bl, bh), (short)0, acc, false, false);
  }

  gemm_epilogue(mode, acc, m0 + (hi ? 8 : 0), n0 + r16,
                outf, outh, ldo, bias, res, dout);
}

// ---------------- causal depthwise conv (width 4) + silu --------------------
__global__ void k_conv(const float* __restrict__ xz, const float* __restrict__ cw,
                       const float* __restrict__ cb,
                       float* __restrict__ xcf, _Float16* __restrict__ xch) {
  int idx = blockIdx.x * 256 + threadIdx.x;      // < 12288*256
  int d = idx & 255; int t = idx >> 8;
  int l = t & 4095; size_t nb = (size_t)(t & ~4095);
  float acc = cb[d];
  for (int k = 0; k < 4; ++k) {
    int lp = l - 3 + k;
    if (lp >= 0) acc += cw[d * 4 + k] * xz[(nb + (size_t)lp) * 512 + d];
  }
  float sv = acc / (1.f + __expf(-acc));         // silu
  xcf[(size_t)t * 256 + d] = sv;
  xch[(size_t)t * 256 + d] = (_Float16)sv;
}

// ------------------- dt projection (K=8, VALU) + softplus -------------------
__global__ void k_dt(const float* __restrict__ xdbl, const float* __restrict__ wdt,
                     const float* __restrict__ bdt, float* __restrict__ delta) {
  int idx = blockIdx.x * 256 + threadIdx.x;      // < 12288*256
  int d = idx & 255; int t = idx >> 8;
  float acc = bdt[d];
  for (int r = 0; r < 8; ++r) acc += xdbl[(size_t)t * 48 + r] * wdt[d * 8 + r];
  delta[(size_t)t * 256 + d] = (acc > 20.f) ? acc : log1pf(__expf(acc));
}

// ---------------------- selective scan, one lane per state ------------------
// 16 states of one (dir,d_inner) channel live in a 16-lane group; the per-step
// y = <h, C_t> reduction is 4 shfl_xor's. Lane s==0 applies +u*D and silu(z)
// gating and stores y in f16 for the out_proj WMMA.
__global__ void k_scan(const float* __restrict__ delta, const float* __restrict__ u,
                       const float* __restrict__ xdbl,  const float* __restrict__ xz,
                       const float* __restrict__ Aneg,  const float* __restrict__ Dv,
                       _Float16* __restrict__ y) {
  int grp = threadIdx.x >> 4;                    // 16 channels per block
  int s = threadIdx.x & 15;                      // state index
  int ch = blockIdx.x * 16 + grp;                // 0..767
  int n = ch >> 8, d = ch & 255;
  float As = Aneg[d * 16 + s];
  float Dp = Dv[d];
  float h = 0.f;
  size_t base = (size_t)n * 4096;
  for (int l = 0; l < 4096; ++l) {
    size_t t = base + l;
    float dl = delta[t * 256 + d];
    float ul = u[t * 256 + d];
    float bm = xdbl[t * 48 + 8 + s];
    float cm = xdbl[t * 48 + 24 + s];
    h = __expf(dl * As) * h + dl * bm * ul;
    float p = h * cm;
    p += __shfl_xor(p, 1, 32);
    p += __shfl_xor(p, 2, 32);
    p += __shfl_xor(p, 4, 32);
    p += __shfl_xor(p, 8, 32);
    if (s == 0) {
      float z = xz[t * 512 + 256 + d];
      float yv = (p + ul * Dp) * (z / (1.f + __expf(-z)));
      y[t * 256 + d] = (_Float16)yv;
    }
  }
}

// ---------------- LN over channels of (s,128) token-major f32 ---------------
__global__ void k_ln2(const float* __restrict__ res,
                      const float* __restrict__ lnw, const float* __restrict__ lnb,
                      _Float16* __restrict__ out) {
  int wid = threadIdx.x >> 5, lane = threadIdx.x & 31;
  int s = blockIdx.x * 8 + wid;
  float v[4];
  float sum = 0.f;
  for (int j = 0; j < 4; ++j) { v[j] = res[(size_t)s * 128 + lane * 4 + j]; sum += v[j]; }
  float mean = wave_sum32(sum) * (1.f / 128.f);
  float var = 0.f;
  for (int j = 0; j < 4; ++j) { float dd = v[j] - mean; var += dd * dd; }
  var = wave_sum32(var) * (1.f / 128.f);
  float r = rsqrtf(var + 1e-6f);
  for (int j = 0; j < 4; ++j) {
    int c = lane * 4 + j;
    out[(size_t)s * 128 + c] = (_Float16)(lnw[c] * ((v[j] - mean) * r) + lnb[c]);
  }
}

// ---------------------------------------------------------------------------
extern "C" void kernel_launch(void* const* d_in, const int* in_sizes, int n_in,
                              void* d_out, int out_size, void* d_ws, size_t ws_size,
                              hipStream_t stream) {
  (void)in_sizes; (void)n_in; (void)out_size; (void)ws_size;
  const float* x        = (const float*)d_in[0];
  const float* ln_w     = (const float*)d_in[1];
  const float* ln_b     = (const float*)d_in[2];
  const float* mnorm_w  = (const float*)d_in[3];
  const float* mnorm_b  = (const float*)d_in[4];
  const float* in_proj  = (const float*)d_in[5];
  const float* conv_w   = (const float*)d_in[6];
  const float* conv_b   = (const float*)d_in[7];
  const float* x_proj   = (const float*)d_in[8];
  const float* dt_w     = (const float*)d_in[9];
  const float* dt_b     = (const float*)d_in[10];
  const float* A_log    = (const float*)d_in[11];
  const float* D_param  = (const float*)d_in[12];
  const float* out_proj = (const float*)d_in[13];
  const float* proj_w   = (const float*)d_in[14];
  const float* proj_b   = (const float*)d_in[15];
  const float* fc1_w    = (const float*)d_in[16];
  const float* fc1_b    = (const float*)d_in[17];
  const float* fc2_w    = (const float*)d_in[18];
  const float* fc2_b    = (const float*)d_in[19];
  float* dout = (float*)d_out;

  char* ws = (char*)d_ws;
  size_t off = 0;
  auto alloc = [&](size_t bytes) -> char* {
    char* p = ws + off;
    off = (off + bytes + 255) & ~(size_t)255;
    return p;
  };
  _Float16* w_inproj16  = (_Float16*)alloc(512 * 128 * 2);
  _Float16* w_xproj16   = (_Float16*)alloc(48 * 256 * 2);      // padded to N=48
  _Float16* w_outproj16 = (_Float16*)alloc(128 * 256 * 2);
  _Float16* w_proj16    = (_Float16*)alloc(128 * 384 * 2);
  _Float16* w_fc1_16    = (_Float16*)alloc(512 * 128 * 2);
  _Float16* w_fc2_16    = (_Float16*)alloc(128 * 512 * 2);
  float*    Aneg        = (float*)   alloc(256 * 16 * 4);
  _Float16* seq16       = (_Float16*)alloc((size_t)12288 * 128 * 2);
  float*    xz          = (float*)   alloc((size_t)12288 * 512 * 4);
  float*    xcf         = (float*)   alloc((size_t)12288 * 256 * 4);
  _Float16* xch         = (_Float16*)alloc((size_t)12288 * 256 * 2);
  float*    xdbl        = (float*)   alloc((size_t)12288 * 48 * 4);
  float*    delta       = (float*)   alloc((size_t)12288 * 256 * 4);
  _Float16* y16         = (_Float16*)alloc((size_t)12288 * 256 * 2);
  _Float16* cat16       = (_Float16*)alloc((size_t)4096 * 384 * 2);
  float*    out_res     = (float*)   alloc((size_t)4096 * 128 * 4);
  _Float16* hln16       = (_Float16*)alloc((size_t)4096 * 128 * 2);
  _Float16* h1_16       = (_Float16*)alloc((size_t)4096 * 512 * 2);

  // --- weight prep (deterministic every call; ~0.3M elements total) ---
  k_f2h<<<256, 256, 0, stream>>>(in_proj,  w_inproj16,  512 * 128);
  k_f2h_pad48<<<48, 256, 0, stream>>>(x_proj, w_xproj16);
  k_f2h<<<128, 256, 0, stream>>>(out_proj, w_outproj16, 128 * 256);
  k_f2h<<<192, 256, 0, stream>>>(proj_w,   w_proj16,    128 * 384);
  k_f2h<<<256, 256, 0, stream>>>(fc1_w,    w_fc1_16,    512 * 128);
  k_f2h<<<256, 256, 0, stream>>>(fc2_w,    w_fc2_16,    128 * 512);
  k_negexp<<<16, 256, 0, stream>>>(A_log, Aneg, 256 * 16);

  // --- LN + LN + 3-way permuted sequences (f16) ---
  k_norm_seq<<<512, 256, 0, stream>>>(x, ln_w, ln_b, mnorm_w, mnorm_b, seq16);

  // --- xz = seq @ in_proj^T : (12288,512), K=128, 16x64 tiles ---
  k_gemm4<<<(768 * 8) / 8, 256, 0, stream>>>(seq16, 128, w_inproj16, 128,
      768, 32, 128, xz, nullptr, 512, EP_F32, nullptr, nullptr, nullptr);

  // --- causal conv4 + silu -> xc (f32 for scan, f16 for GEMMs) ---
  k_conv<<<12288, 256, 0, stream>>>(xz, conv_w, conv_b, xcf, xch);

  // --- x_dbl = xc @ x_proj^T : (12288,48), K=256, single tiles (Nt=3) ---
  k_gemm<<<(768 * 3) / 8, 256, 0, stream>>>(xch, 256, w_xproj16, 256,
      768, 3, 256, xdbl, nullptr, 48, EP_F32, nullptr, nullptr, nullptr);

  // --- delta = softplus(dt @ dt_proj^T + b) : K=8 on VALU ---
  k_dt<<<12288, 256, 0, stream>>>(xdbl, dt_w, dt_b, delta);

  // --- selective scan + u*D + silu(z) gating -> y16 ---
  k_scan<<<48, 256, 0, stream>>>(delta, xcf, xdbl, xz, Aneg, D_param, y16);

  // --- y @ out_proj^T, scattered straight into cat(s, n*128+c) f16 ---
  k_gemm4<<<(768 * 2) / 8, 256, 0, stream>>>(y16, 256, w_outproj16, 256,
      768, 8, 256, nullptr, cat16, 384, EP_CAT16, nullptr, nullptr, nullptr);

  // --- out_res = cat @ proj^T + proj_b + x : (4096,128), K=384 ---
  k_gemm4<<<(256 * 2) / 8, 256, 0, stream>>>(cat16, 384, w_proj16, 384,
      256, 8, 384, out_res, nullptr, 128, EP_BIAS_RES, proj_b, x, nullptr);

  // --- LN(out_res) -> f16 ---
  k_ln2<<<512, 256, 0, stream>>>(out_res, ln_w, ln_b, hln16);

  // --- h1 = gelu(hln @ fc1^T + fc1_b) : (4096,512), K=128 ---
  k_gemm4<<<(256 * 8) / 8, 256, 0, stream>>>(hln16, 128, w_fc1_16, 128,
      256, 32, 128, nullptr, h1_16, 512, EP_BIAS_GELU, fc1_b, nullptr, nullptr);

  // --- out = h1 @ fc2^T + fc2_b + out_res, stored channel-major ---
  k_gemm4<<<(256 * 2) / 8, 256, 0, stream>>>(h1_16, 512, w_fc2_16, 512,
      256, 8, 512, nullptr, nullptr, 0, EP_FINAL, fc2_b, out_res, dout);
}
